// Qwen3MoEMLP_37426345017543
// MI455X (gfx1250) — compile-verified, matrix operand order checked by reading, not run on hardware
//
#include <hip/hip_runtime.h>
#include <math.h>

// ---------------------------------------------------------------------------
// Qwen3 MoE MLP for gfx1250 (MI455X).  T=1024 tokens, H=2048, E=64 experts,
// I=768 intermediate, top-8 routing.  Sparse expert dispatch + bf16 WMMA
// (v_wmma_f32_16x16x32_bf16, fp32 accumulate).
//   gate/up: MBLK=8 register blocking -> each B fragment feeds 8 WMMAs,
//            weights (1.2 GB fp32) streamed ~once; A panel staged to LDS as
//            bf16 (converted once, not per-wave).
//   down:    MBLK=4, A panel (bf16) staged via GLOBAL_LOAD_ASYNC_TO_LDS_B128.
// ---------------------------------------------------------------------------

#define TDIM    1024
#define HDIM    2048
#define NEXP    64
#define IDIM    768
#define TOPK    8
#define NPAIR   (TDIM * TOPK)
#define MBLK_GU 8                // gate/up M-tiles per wave
#define MBLK_DN 4                // down M-tiles per wave
#define KCHUNK  256              // gate/up A-panel K chunk
#define APADB   264              // padded bf16 row stride, gate/up A panel
#define DPAD    776              // padded bf16 row stride, down A panel

typedef __attribute__((ext_vector_type(16))) __bf16 v16bf;
typedef __attribute__((ext_vector_type(8)))  float  v8f;
typedef int v4i_vs __attribute__((vector_size(16)));   // <4 x i32>

#define AS1 __attribute__((address_space(1)))
#define AS3 __attribute__((address_space(3)))

#if __has_builtin(__builtin_amdgcn_global_load_async_to_lds_b128)
#define HAVE_ASYNC_LDS 1
#else
#define HAVE_ASYNC_LDS 0
#endif

// 16-byte global->LDS copy: async (ASYNCcnt-tracked) when available.
static __device__ inline void cp16_lds(const void* g, void* lds) {
#if HAVE_ASYNC_LDS
    __builtin_amdgcn_global_load_async_to_lds_b128(
        (AS1 v4i_vs*)(__UINTPTR_TYPE__)g,
        (AS3 v4i_vs*)(__UINTPTR_TYPE__)lds, 0, 0);
#else
    *(uint4*)lds = *(const uint4*)g;
#endif
}

static __device__ inline void wait_async_lds() {
#if HAVE_ASYNC_LDS
#if __has_builtin(__builtin_amdgcn_s_wait_asynccnt)
    __builtin_amdgcn_s_wait_asynccnt(0);
#else
    asm volatile("s_wait_asynccnt 0x0" ::: "memory");
#endif
#endif
}

static __device__ inline float fast_rcp(float x) {
#if __has_builtin(__builtin_amdgcn_rcpf)
    return __builtin_amdgcn_rcpf(x);
#else
    return 1.f / x;
#endif
}

union FragAB {
    v16bf v;
    uint4 q[2];
};

static __device__ inline v8f wmma_bf16(const v16bf& a, const v16bf& b, const v8f& c) {
    return __builtin_amdgcn_wmma_f32_16x16x32_bf16(false, a, false, b, (short)0, c,
                                                   false, false);
}

// ---------------------------------------------------------------------------
// K0: zero the per-expert counters.
// ---------------------------------------------------------------------------
__global__ void moe_zero_cnt_kernel(int* cnt) {
    if (threadIdx.x < NEXP) cnt[threadIdx.x] = 0;
}

// ---------------------------------------------------------------------------
// K1: router (tiny: 268 MFLOP).  One block / token; thread e -> logit e.
// ---------------------------------------------------------------------------
__global__ void moe_router_kernel(const float* __restrict__ x,
                                  const float* __restrict__ rw,
                                  int* __restrict__ topk_idx,
                                  float* __restrict__ topk_w,
                                  int* __restrict__ cnt) {
    const int t = blockIdx.x;
    const int e = threadIdx.x;

    const float4* xr = (const float4*)(x + (size_t)t * HDIM);
    const float4* wr = (const float4*)(rw + (size_t)e * HDIM);
    float acc = 0.f;
#pragma unroll 4
    for (int h4 = 0; h4 < HDIM / 4; ++h4) {
        float4 a = xr[h4];
        float4 b = wr[h4];
        acc += a.x * b.x + a.y * b.y + a.z * b.z + a.w * b.w;
    }

    __shared__ float sl[NEXP];
    sl[e] = acc;
    __syncthreads();

    if (e == 0) {
        float mx = sl[0];
        for (int i = 1; i < NEXP; ++i) mx = fmaxf(mx, sl[i]);
        float p[NEXP];
        float sum = 0.f;
        for (int i = 0; i < NEXP; ++i) { p[i] = __expf(sl[i] - mx); sum += p[i]; }

        int   idx[TOPK];
        float w[TOPK];
        float wsum = 0.f;
        for (int k = 0; k < TOPK; ++k) {
            int   bj = 0;
            float bv = -1.f;
            for (int i = 0; i < NEXP; ++i)
                if (p[i] > bv) { bv = p[i]; bj = i; }
            idx[k] = bj;
            w[k]   = bv / sum;
            wsum  += w[k];
            p[bj]  = -2.f;
        }
        const float inv = 1.f / wsum;
        for (int k = 0; k < TOPK; ++k) {
            topk_idx[t * TOPK + k] = idx[k];
            topk_w[t * TOPK + k]   = w[k] * inv;
            atomicAdd(&cnt[idx[k]], 1);
        }
    }
}

// ---------------------------------------------------------------------------
// K2: exclusive scan of counts + zero the fill counters.
// ---------------------------------------------------------------------------
__global__ void moe_scan_kernel(const int* __restrict__ cnt,
                                int* __restrict__ off,
                                int* __restrict__ cnt2) {
    if (threadIdx.x < NEXP) cnt2[threadIdx.x] = 0;
    if (threadIdx.x == 0) {
        int run = 0;
        for (int e = 0; e < NEXP; ++e) { off[e] = run; run += cnt[e]; }
    }
}

// ---------------------------------------------------------------------------
// K3: fill per-expert token lists.
// ---------------------------------------------------------------------------
__global__ void moe_fill_kernel(const int* __restrict__ topk_idx,
                                const float* __restrict__ topk_w,
                                const int* __restrict__ off,
                                int* __restrict__ cnt2,
                                int* __restrict__ pair_tok,
                                float* __restrict__ pair_w) {
    const int g = blockIdx.x * blockDim.x + threadIdx.x;
    if (g >= NPAIR) return;
    const int t   = g >> 3;
    const int e   = topk_idx[g];
    const int pos = atomicAdd(&cnt2[e], 1);
    const int p   = off[e] + pos;
    pair_tok[p] = t;
    pair_w[p]   = topk_w[g];
}

// ---------------------------------------------------------------------------
// K4: zero the output (atomic accumulation target).
// ---------------------------------------------------------------------------
__global__ void moe_zero_out_kernel(float4* __restrict__ out) {
    const int g = blockIdx.x * blockDim.x + threadIdx.x;
    out[g] = make_float4(0.f, 0.f, 0.f, 0.f);
}

// ---------------------------------------------------------------------------
// K5: gate/up + SiLU.  grid=(NEXP, IDIM/128), block=256 (8 waves).
// Per 128-row M-group: stage A (gathered token rows) into LDS as bf16 in
// 256-wide K chunks (fp32 load + cvt_pk + ds_store, converted ONCE); inner
// k-loop: 2 B fragments (fp32->bf16 in-register) each feeding 8 WMMAs, A
// fragments are pure ds_load_b128.
// ---------------------------------------------------------------------------
__global__ void __launch_bounds__(256)
moe_gateup_kernel(const float* __restrict__ x,
                  const float* __restrict__ gate_w,
                  const float* __restrict__ up_w,
                  const int* __restrict__ cnt,
                  const int* __restrict__ off,
                  const int* __restrict__ pair_tok,
                  __bf16* __restrict__ hidden) {
    __shared__ __align__(16) __bf16 Asb[128 * APADB];  // ~67.6 KB
    __shared__ int stok[128];

    const int e     = blockIdx.x;
    const int count = cnt[e];
    if (count == 0) return;
    const int wave  = threadIdx.x >> 5;
    const int lane  = threadIdx.x & 31;
    const int base    = off[e];
    const int mgroups = (count + 127) >> 7;
    const int ntbase  = (blockIdx.y * 8 + wave) * 16;     // < IDIM

    const float* gw = gate_w + (size_t)e * IDIM * HDIM;
    const float* uw = up_w   + (size_t)e * IDIM * HDIM;

    const int r_frag  = lane & 15;
    const int c0_frag = (lane >> 4) * 8;

    for (int mg = 0; mg < mgroups; ++mg) {
        if (threadIdx.x < 128) {
            int m = mg * 128 + (int)threadIdx.x;
            if (m >= count) m = count - 1;            // clamp: rows discarded later
            stok[threadIdx.x] = pair_tok[base + m];
        }

        v8f cg[MBLK_GU], cu[MBLK_GU];
#pragma unroll
        for (int mb = 0; mb < MBLK_GU; ++mb) {
            cg[mb] = (v8f){0.f, 0.f, 0.f, 0.f, 0.f, 0.f, 0.f, 0.f};
            cu[mb] = (v8f){0.f, 0.f, 0.f, 0.f, 0.f, 0.f, 0.f, 0.f};
        }

        for (int kc = 0; kc < HDIM; kc += KCHUNK) {
            __syncthreads();   // previous chunk fully consumed / stok visible
            // stage 128 x KCHUNK: fp32 float4 load -> 4x bf16 -> ds_store_b64
#pragma unroll
            for (int it = 0; it < 32; ++it) {
                const int idx = (int)threadIdx.x + it * 256;
                const int r   = idx >> 6;             // 64 float4-chunks per row
                const int c4  = idx & 63;
                float4 v = *(const float4*)(x + (size_t)stok[r] * HDIM + kc + c4 * 4);
                union { __bf16 h[4]; uint2 u2; } pk;
                pk.h[0] = (__bf16)v.x; pk.h[1] = (__bf16)v.y;
                pk.h[2] = (__bf16)v.z; pk.h[3] = (__bf16)v.w;
                *(uint2*)&Asb[r * APADB + c4 * 4] = pk.u2;
            }
            __syncthreads();

            for (int ks = 0; ks < KCHUNK; ks += 32) {
                // B fragments from the contiguous fp32 weight rows
                const int i_row = ntbase + r_frag;
                const float* gp = gw + (size_t)i_row * HDIM + kc + ks + c0_frag;
                const float* up = uw + (size_t)i_row * HDIM + kc + ks + c0_frag;
                __builtin_prefetch(gp + 32, 0, 0);
                __builtin_prefetch(up + 32, 0, 0);
                FragAB fg, fu;
#pragma unroll
                for (int j = 0; j < 8; ++j) {
                    fg.v[j]     = (__bf16)gp[j];
                    fg.v[8 + j] = (__bf16)gp[16 + j];
                    fu.v[j]     = (__bf16)up[j];
                    fu.v[8 + j] = (__bf16)up[16 + j];
                }

                // 8 M-tiles reuse each B fragment; A is pure ds_load_b128
#pragma unroll
                for (int mb = 0; mb < MBLK_GU; ++mb) {
                    FragAB fa;
                    const __bf16* ap = &Asb[(mb * 16 + r_frag) * APADB + ks + c0_frag];
                    fa.q[0] = *(const uint4*)(ap);
                    fa.q[1] = *(const uint4*)(ap + 16);
                    cg[mb] = wmma_bf16(fa.v, fg.v, cg[mb]);
                    cu[mb] = wmma_bf16(fa.v, fu.v, cu[mb]);
                }
            }
        }

        // epilogue: h = silu(gate) * up  ->  bf16 hidden
        const int n = ntbase + r_frag;
#pragma unroll
        for (int mb = 0; mb < MBLK_GU; ++mb) {
            const int mrow0 = mg * 128 + mb * 16 + c0_frag;
#pragma unroll
            for (int r = 0; r < 8; ++r) {
                const int m = mrow0 + r;
                if (m < count) {
                    const float g = cg[mb][r];
                    const float u = cu[mb][r];
                    const float h = g * fast_rcp(1.f + __expf(-g)) * u;
                    hidden[(size_t)(base + m) * IDIM + n] = (__bf16)h;
                }
            }
        }
    }
}

// ---------------------------------------------------------------------------
// K6: down projection + weighted atomic scatter.
// grid=(NEXP, HDIM/128), block=256.  A panel (64 x 768 bf16, padded) staged
// once per M-group via async global->LDS (pure byte copy); inner k-loop:
// one B fragment -> 4 WMMAs.
// ---------------------------------------------------------------------------
__global__ void __launch_bounds__(256)
moe_down_kernel(const __bf16* __restrict__ hidden,
                const float* __restrict__ down_w,
                const int* __restrict__ cnt,
                const int* __restrict__ off,
                const int* __restrict__ pair_tok,
                const float* __restrict__ pair_w,
                float* __restrict__ out) {
    __shared__ __align__(16) __bf16 Ash[64 * DPAD];    // ~97 KB

    const int e     = blockIdx.x;
    const int count = cnt[e];
    if (count == 0) return;
    const int wave  = threadIdx.x >> 5;
    const int lane  = threadIdx.x & 31;
    const int base    = off[e];
    const int mgroups = (count + 63) >> 6;
    const int ntbase  = (blockIdx.y * 8 + wave) * 16;     // < HDIM

    const float* dw = down_w + (size_t)e * HDIM * IDIM;

    const int r_frag  = lane & 15;
    const int c0_frag = (lane >> 4) * 8;

    for (int mg = 0; mg < mgroups; ++mg) {
        __syncthreads();   // previous group consumed
        // stage 64 x IDIM bf16: 6144 x 16B chunks, 24 per thread (async)
#pragma unroll
        for (int it = 0; it < 24; ++it) {
            const int idx = (int)threadIdx.x + it * 256;
            const int r   = idx / 96;                 // 96 chunks per row
            const int c   = idx - r * 96;             // 8 bf16 per chunk
            int m = mg * 64 + r;
            if (m >= count) m = count - 1;            // clamp: rows discarded
            cp16_lds(hidden + (size_t)(base + m) * IDIM + c * 8,
                     &Ash[r * DPAD + c * 8]);
        }
        wait_async_lds();
        __syncthreads();

        v8f c[MBLK_DN];
#pragma unroll
        for (int mb = 0; mb < MBLK_DN; ++mb)
            c[mb] = (v8f){0.f, 0.f, 0.f, 0.f, 0.f, 0.f, 0.f, 0.f};

        for (int ks = 0; ks < IDIM; ks += 32) {
            const int h_row = ntbase + r_frag;
            const float* dp = dw + (size_t)h_row * IDIM + ks + c0_frag;
            __builtin_prefetch(dp + 32, 0, 0);
            FragAB fb;
#pragma unroll
            for (int j = 0; j < 8; ++j) {
                fb.v[j]     = (__bf16)dp[j];
                fb.v[8 + j] = (__bf16)dp[16 + j];
            }

#pragma unroll
            for (int mb = 0; mb < MBLK_DN; ++mb) {
                FragAB fa;
                const __bf16* ap = &Ash[(mb * 16 + r_frag) * DPAD + ks + c0_frag];
                fa.q[0] = *(const uint4*)(ap);
                fa.q[1] = *(const uint4*)(ap + 16);
                c[mb] = wmma_bf16(fa.v, fb.v, c[mb]);
            }
        }

        const int n = ntbase + r_frag;
#pragma unroll
        for (int mb = 0; mb < MBLK_DN; ++mb) {
            const int mrow0 = mg * 64 + mb * 16 + c0_frag;
#pragma unroll
            for (int r = 0; r < 8; ++r) {
                const int m = mrow0 + r;
                if (m < count) {
                    const int   t = pair_tok[base + m];
                    const float w = pair_w[base + m];
                    atomicAdd(&out[(size_t)t * HDIM + n], w * c[mb][r]);
                }
            }
        }
    }
}

// ---------------------------------------------------------------------------
// Launch
// ---------------------------------------------------------------------------
extern "C" void kernel_launch(void* const* d_in, const int* in_sizes, int n_in,
                              void* d_out, int out_size, void* d_ws, size_t ws_size,
                              hipStream_t stream) {
    const float* x        = (const float*)d_in[0];   // [T, H]
    const float* router_w = (const float*)d_in[1];   // [E, H]
    const float* gate_w   = (const float*)d_in[2];   // [E, I, H]
    const float* up_w     = (const float*)d_in[3];   // [E, I, H]
    const float* down_w   = (const float*)d_in[4];   // [E, H, I]
    float*       out      = (float*)d_out;           // [T, H]

    char* ws = (char*)d_ws;
    int*    cnt      = (int*)(ws + 0);
    int*    cnt2     = (int*)(ws + 256);
    int*    off      = (int*)(ws + 512);
    int*    pair_tok = (int*)(ws + 1024);
    float*  pair_w   = (float*)(ws + 1024 + 32768);
    int*    topk_idx = (int*)(ws + 1024 + 65536);
    float*  topk_w   = (float*)(ws + 1024 + 98304);
    __bf16* hidden   = (__bf16*)(ws + 132352);        // 16B aligned, 12.6 MB

    moe_zero_cnt_kernel<<<1, 64, 0, stream>>>(cnt);
    moe_router_kernel<<<TDIM, 64, 0, stream>>>(x, router_w, topk_idx, topk_w, cnt);
    moe_scan_kernel<<<1, 64, 0, stream>>>(cnt, off, cnt2);
    moe_fill_kernel<<<NPAIR / 256, 256, 0, stream>>>(topk_idx, topk_w, off, cnt2,
                                                     pair_tok, pair_w);
    moe_zero_out_kernel<<<(TDIM * HDIM / 4) / 256, 256, 0, stream>>>((float4*)out);
    moe_gateup_kernel<<<dim3(NEXP, IDIM / 128), 256, 0, stream>>>(
        x, gate_w, up_w, cnt, off, pair_tok, hidden);
    moe_down_kernel<<<dim3(NEXP, HDIM / 128), 256, 0, stream>>>(
        hidden, down_w, cnt, off, pair_tok, pair_w, out);
    (void)in_sizes; (void)n_in; (void)out_size; (void)ws_size;
}
